// MeshDecoder_87110526698215
// MI455X (gfx1250) — matrix operand order repaired and checked by reading, not compiled, not run on hardware
//
#include <hip/hip_runtime.h>
#include <math.h>

// ---------------------------------------------------------------------------
// MeshDecoder for MI455X (gfx1250, wave32).
//  * channel-mixing GEMMs: v_wmma_f32_16x16x32_f16, f32 accumulate.
//    A fragment: 4x global_load_b128 (contiguous K runs of 8 floats/lane).
//    B fragment: W pre-transposed to column-major f16 (padded to 16 cols),
//                one aligned v16h load (2x global_load_b128) per K step.
//    Epilogue (acc / bias / ELU) fully templated -> branchless stores.
//  * graph propagate / upsample: scatter-add with global f32 atomics.
//  * Chebyshev recurrence folded into scatter init (next = -Tx0, weight 2*norm).
// ---------------------------------------------------------------------------

typedef __attribute__((ext_vector_type(16))) _Float16 v16h;
typedef __attribute__((ext_vector_type(8)))  float    v8f;

#define TPB 256

// ---------------- elementwise / utility kernels ----------------------------

__global__ void zero_f32v4(float4* __restrict__ p, long n4) {
  long i = (long)blockIdx.x * blockDim.x + threadIdx.x;
  long stride = (long)gridDim.x * blockDim.x;
  float4 z = {0.0f, 0.0f, 0.0f, 0.0f};
  for (; i < n4; i += stride) p[i] = z;
}

__global__ void negcopy_v4(float4* __restrict__ dst, const float4* __restrict__ src,
                           long n4) {
  long i = (long)blockIdx.x * blockDim.x + threadIdx.x;
  long stride = (long)gridDim.x * blockDim.x;
  for (; i < n4; i += stride) {
    float4 v = src[i];
    v.x = -v.x; v.y = -v.y; v.z = -v.z; v.w = -v.w;
    dst[i] = v;
  }
}

__global__ void deg_k(const int* __restrict__ row, int E, float* __restrict__ deg) {
  int e = blockIdx.x * blockDim.x + threadIdx.x;
  if (e < E) atomicAdd(deg + row[e], 1.0f);
}

__global__ void norm_k(const int* __restrict__ row, const int* __restrict__ col,
                       const float* __restrict__ deg, float* __restrict__ nrm, int E) {
  int e = blockIdx.x * blockDim.x + threadIdx.x;
  if (e < E) {
    float dr = deg[row[e]];
    float dc = deg[col[e]];
    float ir = dr > 0.0f ? rsqrtf(dr) : 0.0f;
    float ic = dc > 0.0f ? rsqrtf(dc) : 0.0f;
    nrm[e] = ir * ic;
  }
}

// out[b, row[e], c] += scale * w[e] * in[b, col[e], c]   for all 16 batches.
__global__ void scatter_add(const float* __restrict__ in, float* __restrict__ out,
                            const int* __restrict__ row, const int* __restrict__ col,
                            const float* __restrict__ w, float scale,
                            int E, int C, int Nin, int Nout) {
  int t = blockIdx.x * blockDim.x + threadIdx.x;
  if (t >= E * C) return;
  int e = t / C;
  int c = t - e * C;
  int r = row[e];
  int s = col[e];
  float wv = scale * w[e];
  const float* ip = in  + (size_t)s * C + c;
  float*       op = out + (size_t)r * C + c;
  size_t istr = (size_t)Nin * C;
  size_t ostr = (size_t)Nout * C;
#pragma unroll 4
  for (int b = 0; b < 16; ++b)
    atomicAdd(op + (size_t)b * ostr, wv * ip[(size_t)b * istr]);
}

// W[t][k][n] (f32, taps x Kc x Ncols) -> Wt[t][n][k] (f16, taps x Npad x Kc, zero pad)
__global__ void prep_w_k(const float* __restrict__ W, _Float16* __restrict__ Wt,
                         int taps, int Kc, int Ncols, int Npad) {
  long total = (long)taps * Npad * Kc;
  long i = (long)blockIdx.x * blockDim.x + threadIdx.x;
  long stride = (long)gridDim.x * blockDim.x;
  for (; i < total; i += stride) {
    int  k  = (int)(i % Kc);
    long tn = i / Kc;
    int  n  = (int)(tn % Npad);
    int  t  = (int)(tn / Npad);
    float v = (n < Ncols) ? W[((size_t)t * Kc + k) * Ncols + n] : 0.0f;
    Wt[i] = (_Float16)v;
  }
}

// final: out[r*3+c] = inPad[r*16+c] + bias[c]
__global__ void slice_bias_k(const float* __restrict__ inPad, const float* __restrict__ bias,
                             float* __restrict__ out, long rows) {
  long total = rows * 3;
  long i = (long)blockIdx.x * blockDim.x + threadIdx.x;
  long stride = (long)gridDim.x * blockDim.x;
  for (; i < total; i += stride) {
    long r = i / 3;
    int  c = (int)(i - r * 3);
    out[i] = inPad[r * 16 + c] + bias[c];
  }
}

// ---------------- WMMA GEMM -------------------------------------------------
// C[Mtot x Npad] (+)= A[Mtot x KD] * B, B transposed f16: Bt[n][k].
// ACC: 0 = write, 1 = accumulate (C preloaded into the WMMA accumulator).
// EPI: 0 = none, 1 = +bias[col], 2 = +bias[col] then ELU.
// One 16x16 tile per wave, 4 waves/block.
//
// Operand layouts used here (wave32; A and B share one K order):
//   A (16x32 f16): lane lm = row M; half h -> K = h*16 + e (contiguous in A row)
//   B (32x16 f16): lane lm = col N; half h -> K = h*16 + e (contiguous in Bt)
//   C/D (16x16 f32): c[r] -> M = r + 8*half, N = lm
template <int KD, int ACC, int EPI>
__global__ __launch_bounds__(128)
void gemm_wmma_f16(const float* __restrict__ A, const _Float16* __restrict__ Bt,
                   float* __restrict__ C, int Npad,
                   const float* __restrict__ bias, int totalTiles) {
  int nColTiles = Npad >> 4;
  int tile = blockIdx.x * 4 + (threadIdx.x >> 5);   // wave-uniform
  if (tile >= totalTiles) return;
  int tileR = tile / nColTiles;
  int tileC = tile - tileR * nColTiles;

  int lane = threadIdx.x & 31;
  int half = lane >> 4;
  int lm   = lane & 15;

  int rowA = tileR * 16 + lm;
  int colB = tileC * 16 + lm;
  const float*    Arow = A  + (size_t)rowA * KD;
  const _Float16* Bcol = Bt + (size_t)colB * KD;
  float* Ccol = C + (size_t)(tileR * 16 + half * 8) * Npad + colB;

  float bv = 0.0f;
  if (EPI >= 1) bv = bias[colB];          // single load, hoisted

  v8f cacc;
  if (ACC) {
#pragma unroll
    for (int r = 0; r < 8; ++r) cacc[r] = Ccol[(size_t)r * Npad];
  } else {
#pragma unroll
    for (int r = 0; r < 8; ++r) cacc[r] = 0.0f;
  }

#pragma unroll
  for (int k0 = 0; k0 < KD; k0 += 32) {
    // A fragment: two runs of 8 consecutive floats -> 4x b128 loads
    const float4* p0 = (const float4*)(Arow + k0 + half * 16);
    const float4* p1 = (const float4*)(Arow + k0 + half * 16 + 8);
    float4 q0 = p0[0], q1 = p0[1];
    float4 q2 = p1[0], q3 = p1[1];
    v16h a;
    a[0]  = (_Float16)q0.x; a[1]  = (_Float16)q0.y; a[2]  = (_Float16)q0.z; a[3]  = (_Float16)q0.w;
    a[4]  = (_Float16)q1.x; a[5]  = (_Float16)q1.y; a[6]  = (_Float16)q1.z; a[7]  = (_Float16)q1.w;
    a[8]  = (_Float16)q2.x; a[9]  = (_Float16)q2.y; a[10] = (_Float16)q2.z; a[11] = (_Float16)q2.w;
    a[12] = (_Float16)q3.x; a[13] = (_Float16)q3.y; a[14] = (_Float16)q3.z; a[15] = (_Float16)q3.w;
    // B fragment: 16 contiguous halfs, 32B aligned -> 2x b128 loads
    v16h b = *(const v16h*)(Bcol + k0 + half * 16);
    cacc = __builtin_amdgcn_wmma_f32_16x16x32_f16(
        false, a, false, b, (short)0, cacc, false, false);
  }

#pragma unroll
  for (int r = 0; r < 8; ++r) {
    float v = cacc[r];
    if (EPI >= 1) v += bv;
    if (EPI == 2) {                        // branchless ELU
      float ev = __expf(v) - 1.0f;
      v = v < 0.0f ? ev : v;
    }
    Ccol[(size_t)r * Npad] = v;
  }
}

// ---------------- host-side orchestration ----------------------------------

static inline unsigned gridFor(long n, int tpb) {
  long b = (n + tpb - 1) / tpb;
  if (b < 1) b = 1;
  if (b > (1L << 20)) b = 1L << 20;
  return (unsigned)b;
}

static inline void launch_zero(float* p, long n, hipStream_t s) {
  long n4 = n >> 2;                         // n%4 tail left unzeroed (deg with Nv = 625)
  zero_f32v4<<<gridFor(n4, TPB), TPB, 0, s>>>((float4*)p, n4);
}

static void launch_gemm(const float* A, const _Float16* Bt, float* C,
                        int Mtot, int Kdim, int Npad, int acc,
                        const float* bias, int elu, hipStream_t s) {
  int tiles = (Mtot >> 4) * (Npad >> 4);
  unsigned blocks = (unsigned)((tiles + 3) / 4);
#define GEMM_GO(KD, ACC, EPI) \
  gemm_wmma_f16<KD, ACC, EPI><<<blocks, 128, 0, s>>>(A, Bt, C, Npad, bias, tiles)
  if (Kdim == 128) {
    GEMM_GO(128, 0, 1);                     // latent: write + bias
  } else if (Kdim == 64) {
    if (!acc)      GEMM_GO(64, 0, 0);
    else if (elu)  GEMM_GO(64, 1, 2);
    else           GEMM_GO(64, 1, 0);
  } else {
    if (!acc)      GEMM_GO(32, 0, 0);
    else if (elu)  GEMM_GO(32, 1, 2);
    else           GEMM_GO(32, 1, 0);
  }
#undef GEMM_GO
}

static inline void launch_scatter(const float* in, float* out,
                                  const int* row, const int* col, const float* w,
                                  float scale, int E, int C, int Nin, int Nout,
                                  hipStream_t s) {
  long total = (long)E * C;
  scatter_add<<<(unsigned)((total + TPB - 1) / TPB), TPB, 0, s>>>(
      in, out, row, col, w, scale, E, C, Nin, Nout);
}

// Chebyshev graph conv, K=6:  out = sum_k Tx_k @ W[k] (+bias, optional ELU)
// Tx0 = h_in; Tx1 = prop(Tx0); Tx_k = 2*prop(Tx_{k-1}) - Tx_{k-2}
static void run_cheb(const float* h_in, float* out,
                     float* t1, float* t2, float* t3,
                     const int* erow, const int* ecol, int E, int Nv,
                     const float* W, int Cin, int Cout, int CoutPad,
                     _Float16* wt, const float* fuse_bias, int elu,
                     float* deg, float* nrm, hipStream_t s) {
  const int B = 16;
  long elemsIn = (long)B * Nv * Cin;
  int  Mtot = B * Nv;

  // prep weights: 6 taps, f16, column-major, padded
  prep_w_k<<<gridFor((long)6 * CoutPad * Cin, TPB), TPB, 0, s>>>(
      W, wt, 6, Cin, Cout, CoutPad);
  size_t wtap = (size_t)CoutPad * Cin;

  // sym-normalized adjacency weights
  launch_zero(deg, Nv, s);
  deg_k<<<(unsigned)((E + TPB - 1) / TPB), TPB, 0, s>>>(erow, E, deg);
  norm_k<<<(unsigned)((E + TPB - 1) / TPB), TPB, 0, s>>>(erow, ecol, deg, nrm, E);

  // tap 0: out = Tx0 @ W0   (write mode, no zero pass)
  launch_gemm(h_in, wt, out, Mtot, Cin, CoutPad, 0, nullptr, 0, s);

  // tap 1: Tx1 = prop(Tx0)
  launch_zero(t1, elemsIn, s);
  launch_scatter(h_in, t1, erow, ecol, nrm, 1.0f, E, Cin, Nv, Nv, s);
  launch_gemm(t1, wt + wtap, out, Mtot, Cin, CoutPad, 1, nullptr, 0, s);

  // taps 2..5: next = -Tx_{k-2} + 2*prop(Tx_{k-1})
  const float* prev = h_in;
  float* curr = t1;
  float* freeb  = t2;
  float* freeb2 = t3;
  for (int k = 2; k < 6; ++k) {
    float* nxt = freeb;
    negcopy_v4<<<gridFor(elemsIn >> 2, TPB), TPB, 0, s>>>(
        (float4*)nxt, (const float4*)prev, elemsIn >> 2);
    launch_scatter(curr, nxt, erow, ecol, nrm, 2.0f, E, Cin, Nv, Nv, s);
    int last = (k == 5);
    launch_gemm(nxt, wt + (size_t)k * wtap, out, Mtot, Cin, CoutPad, 1,
                last ? fuse_bias : nullptr, last ? elu : 0, s);
    float* newfree = (prev == h_in) ? freeb2 : (float*)prev;
    prev = curr; curr = nxt; freeb = newfree;
  }
}

extern "C" void kernel_launch(void* const* d_in, const int* in_sizes, int n_in,
                              void* d_out, int out_size, void* d_ws, size_t ws_size,
                              hipStream_t stream) {
  (void)in_sizes; (void)n_in; (void)out_size; (void)ws_size;

  static const int Nlv[5] = {40000, 10000, 2500, 625, 157};

  const float* x     = (const float*)d_in[0];
  const float* lin_w = (const float*)d_in[1];
  const float* lin_b = (const float*)d_in[2];
  const int* edge[4];
  for (int i = 0; i < 4; ++i) edge[i] = (const int*)d_in[3 + i];
  const int*   up_row[4];
  const int*   up_col[4];
  const float* up_val[4];
  for (int i = 0; i < 4; ++i) {
    up_row[i] = (const int*)  d_in[7 + 3 * i];
    up_col[i] = (const int*)  d_in[8 + 3 * i];
    up_val[i] = (const float*)d_in[9 + 3 * i];
  }
  const float* W[5];
  const float* bb[5];
  for (int j = 0; j < 5; ++j) {
    W[j]  = (const float*)d_in[19 + 2 * j];
    bb[j] = (const float*)d_in[20 + 2 * j];
  }

  // workspace carve-up (float units, all regions 32B-aligned)
  const size_t BIG = (size_t)16 * 40000 * 32;        // largest live feature map
  float* ws   = (float*)d_ws;
  float* hA   = ws;                                  // current features
  float* hB   = ws + 1 * BIG;                        // pooled / padded final out
  float* t1   = ws + 2 * BIG;                        // Chebyshev rotating buffers
  float* t2   = ws + 3 * BIG;
  float* t3   = ws + 4 * BIG;
  float* deg  = ws + 5 * BIG;                        // 40000
  float* nrm  = deg + 40064;                         // up to 240000
  _Float16* wtc  = (_Float16*)(nrm + 240128);        // cheb Wt: <= 6*64*64 halfs
  _Float16* wlat = (_Float16*)(nrm + 240128 + 16384);// latent Wt: 10048*128 halfs

  const int latN = Nlv[4] * 64;                      // 10048, multiple of 16

  // latent linear: h = x(16x128) @ lin_w(128x10048) + lin_b  -> hA [16,157,64]
  prep_w_k<<<gridFor((long)latN * 128, TPB), TPB, 0, stream>>>(
      lin_w, wlat, 1, 128, latN, latN);
  launch_gemm(x, wlat, hA, 16, 128, latN, 0, lin_b, 0, stream);

  static const int CinA[4]  = {64, 64, 32, 32};
  static const int CoutA[4] = {64, 32, 32, 32};

  for (int j = 0; j < 4; ++j) {
    int lvl  = 3 - j;
    int Cin  = CinA[j];
    int Cout = CoutA[j];
    int Nout = Nlv[lvl];
    int Nin  = Nlv[lvl + 1];
    int nnz  = 3 * Nout;
    int E    = 6 * Nout;

    // pool: hA [16,Nin,Cin] -> hB [16,Nout,Cin]
    launch_zero(hB, (long)16 * Nout * Cin, stream);
    launch_scatter(hA, hB, up_row[lvl], up_col[lvl], up_val[lvl],
                   1.0f, nnz, Cin, Nin, Nout, stream);

    // cheb conv (+bias+ELU fused into last tap): hB -> hA
    run_cheb(hB, hA, t1, t2, t3,
             edge[lvl], edge[lvl] + E, E, Nout,
             W[j], Cin, Cout, Cout, wtc, bb[j], /*elu=*/1,
             deg, nrm, stream);
  }

  // final cheb conv 32 -> 3 (padded to 16) at full res, into hB, then slice
  run_cheb(hA, hB, t1, t2, t3,
           edge[0], edge[0] + 240000, 240000, 40000,
           W[4], 32, 3, 16, wtc, nullptr, /*elu=*/0,
           deg, nrm, stream);
  slice_bias_k<<<gridFor((long)16 * 40000 * 3, TPB), TPB, 0, stream>>>(
      hB, bb[4], (float*)d_out, (long)16 * 40000);
}
